// GNNLayer_4896262717834
// MI455X (gfx1250) — compile-verified, matrix operand order checked by reading, not run on hardware
//
#include <hip/hip_runtime.h>
#include <hip/hip_bf16.h>

#define HDIM 128
#define WAVES 8
#define TPB 256
#define EDGE_TILES 8   // edge tiles (of 128 edges) per workgroup, amortizes LDS weight staging

typedef __attribute__((ext_vector_type(16))) __bf16 bf16x16;
typedef __attribute__((ext_vector_type(8)))  __bf16 bf16x8;
typedef __attribute__((ext_vector_type(4)))  __bf16 bf16x4;
typedef __attribute__((ext_vector_type(8)))  float  f32x8;
typedef __attribute__((ext_vector_type(4)))  float  f32x4;

static __device__ __forceinline__ float silu_f(float v) {
  return v * (1.0f / (1.0f + __expf(-v)));
}

static __device__ __forceinline__ bf16x4 cvt4(f32x4 v) {
  bf16x4 r;
  r.x = (__bf16)v.x; r.y = (__bf16)v.y; r.z = (__bf16)v.z; r.w = (__bf16)v.w;
  return r;
}

// Load one 16x32 bf16 WMMA fragment from a row-major [16][ldk] bf16 tile.
// Per CDNA5 ISA 7.12.2 (16-bit A 16x32): lanes 0-15 hold M=lane, K={ks..ks+7, ks+16..ks+23};
// lanes 16-31 hold M=lane-16, K={ks+8..ks+15, ks+24..ks+31}. B (stored transposed, lane=N)
// uses the identical addressing.
static __device__ __forceinline__ bf16x16 load_frag(const __bf16* tile, int ldk, int ks, int lane) {
  const int m  = lane & 15;
  const int k0 = ks + ((lane >> 4) << 3);
  const __bf16* p = tile + m * ldk + k0;
  union { bf16x16 v; bf16x8 h[2]; } u;
  u.h[0] = *(const bf16x8*)(p);
  u.h[1] = *(const bf16x8*)(p + 16);
  return u.v;
}

static __device__ __forceinline__ f32x8 wmma_bf16(bf16x16 a, bf16x16 b, f32x8 c) {
  return __builtin_amdgcn_wmma_f32_16x16x32_bf16(false, a, false, b, (short)0, c, false, false);
}

// One wave computes a 16xHDIM output tile: 8 accumulators of 16x16, K-stepped by 32.
static __device__ __forceinline__ void gemm_16xH(const __bf16* Atile, const __bf16* Bt,
                                                 int K, int lane, f32x8 acc[8]) {
#pragma unroll
  for (int nt = 0; nt < 8; ++nt)
    acc[nt] = f32x8{0.f, 0.f, 0.f, 0.f, 0.f, 0.f, 0.f, 0.f};
  for (int ks = 0; ks < K; ks += 32) {
    bf16x16 a = load_frag(Atile, K, ks, lane);
#pragma unroll
    for (int nt = 0; nt < 8; ++nt) {
      bf16x16 b = load_frag(Bt + nt * 16 * K, K, ks, lane);
      acc[nt] = wmma_bf16(a, b, acc[nt]);
    }
  }
}

// ---------------------------------------------------------------- prep kernels
__global__ void zero_kernel(float* __restrict__ p, int n) {
  int i = blockIdx.x * blockDim.x + threadIdx.x;
  if (i < n) p[i] = 0.0f;
}

// W [K][N] f32 row-major -> Wt [N][K] bf16 row-major
__global__ void wconv_kernel(const float* __restrict__ W, __bf16* __restrict__ Wt, int K, int N) {
  int i = blockIdx.x * blockDim.x + threadIdx.x;
  if (i >= K * N) return;
  int k = i / N, n = i - k * N;
  Wt[n * K + k] = (__bf16)W[k * N + n];
}

// ---------------------------------------------------------------- node MLPs
// fm = silu(silu(x@W1+b1)@W2+b2);  m_t = fm @ Mind
__global__ __launch_bounds__(TPB) void node_kernel(
    const float* __restrict__ x, const float* __restrict__ b1, const float* __restrict__ b2,
    const __bf16* __restrict__ Wt1, const __bf16* __restrict__ Wt2, const __bf16* __restrict__ Wti,
    float* __restrict__ fm, float* __restrict__ mt, int nNodes) {
  __shared__ __align__(16) __bf16 sW1[HDIM * HDIM];
  __shared__ __align__(16) __bf16 sW2[HDIM * HDIM];
  __shared__ __align__(16) __bf16 sWi[HDIM * HDIM];
  __shared__ __align__(16) __bf16 sA[WAVES][16 * HDIM];
  __shared__ __align__(16) __bf16 sS[WAVES][16 * HDIM];

  const int tid = threadIdx.x, wave = tid >> 5, lane = tid & 31;
  const int hi = (lane >> 4) << 3, nl = lane & 15;

  {  // stage weights to LDS
    const uint4* s1 = (const uint4*)Wt1; const uint4* s2 = (const uint4*)Wt2;
    const uint4* s3 = (const uint4*)Wti;
    uint4* d1 = (uint4*)sW1; uint4* d2 = (uint4*)sW2; uint4* d3 = (uint4*)sWi;
    for (int i = tid; i < HDIM * HDIM / 8; i += TPB) { d1[i] = s1[i]; d2[i] = s2[i]; d3[i] = s3[i]; }
  }

  const int row0 = blockIdx.x * (WAVES * 16) + wave * 16;
  for (int r = 0; r < 16; ++r) {  // A tile = bf16(x rows)
    int row = row0 + r;
    f32x4 v = {0.f, 0.f, 0.f, 0.f};
    if (row < nNodes) v = *(const f32x4*)(x + (size_t)row * HDIM + lane * 4);
    *(bf16x4*)(&sA[wave][r * HDIM + lane * 4]) = cvt4(v);
  }
  __syncthreads();

  f32x8 acc[8];
  // stage 1: silu(x@W1+b1) -> sS (bf16)
  gemm_16xH(sA[wave], sW1, HDIM, lane, acc);
#pragma unroll
  for (int nt = 0; nt < 8; ++nt) {
    int n = nt * 16 + nl; float bias = b1[n];
#pragma unroll
    for (int r = 0; r < 8; ++r)
      sS[wave][(r + hi) * HDIM + n] = (__bf16)silu_f(acc[nt][r] + bias);
  }
  // stage 2: fm = silu(sS@W2+b2) -> global f32 + sA (bf16)
  gemm_16xH(sS[wave], sW2, HDIM, lane, acc);
#pragma unroll
  for (int nt = 0; nt < 8; ++nt) {
    int n = nt * 16 + nl; float bias = b2[n];
#pragma unroll
    for (int r = 0; r < 8; ++r) {
      int row = row0 + r + hi;
      float v = silu_f(acc[nt][r] + bias);
      sA[wave][(r + hi) * HDIM + n] = (__bf16)v;
      if (row < nNodes) fm[(size_t)row * HDIM + n] = v;
    }
  }
  // stage 3: m_t = fm @ Mind
  gemm_16xH(sA[wave], sWi, HDIM, lane, acc);
#pragma unroll
  for (int nt = 0; nt < 8; ++nt) {
    int n = nt * 16 + nl;
#pragma unroll
    for (int r = 0; r < 8; ++r) {
      int row = row0 + r + hi;
      if (row < nNodes) mt[(size_t)row * HDIM + n] = acc[nt][r];
    }
  }
}

// ---------------------------------------------------------------- edge MLP + scatter
__global__ __launch_bounds__(TPB) void edge_kernel(
    const float* __restrict__ x, const int* __restrict__ eidx, const float* __restrict__ nd,
    const float* __restrict__ est, const float* __restrict__ b1, const float* __restrict__ b2,
    const __bf16* __restrict__ Wt1, const __bf16* __restrict__ Wt2,
    const float* __restrict__ fm, float* __restrict__ seg, float* __restrict__ cnt,
    int nEdges) {
  __shared__ __align__(16) __bf16 sW1[HDIM * 3 * HDIM];     // 96 KB [128][384]
  __shared__ __align__(16) __bf16 sW2[HDIM * HDIM];         // 32 KB
  __shared__ __align__(16) __bf16 sA[WAVES][16 * 3 * HDIM]; // 96 KB
  __shared__ __align__(16) __bf16 sS[WAVES][16 * HDIM];     // 32 KB
  __shared__ int   sSrc[WAVES][16];
  __shared__ int   sDst[WAVES][16];
  __shared__ float sDec[WAVES][16];

  const int tid = threadIdx.x, wave = tid >> 5, lane = tid & 31;
  const int hi = (lane >> 4) << 3, nl = lane & 15;

  {  // stage weights to LDS (amortized over EDGE_TILES tiles)
    const uint4* s1 = (const uint4*)Wt1; const uint4* s2 = (const uint4*)Wt2;
    uint4* d1 = (uint4*)sW1; uint4* d2 = (uint4*)sW2;
    for (int i = tid; i < 3 * HDIM * HDIM / 8; i += TPB) d1[i] = s1[i];
    for (int i = tid; i < HDIM * HDIM / 8; i += TPB) d2[i] = s2[i];
  }
  __syncthreads();

  for (int t = 0; t < EDGE_TILES; ++t) {
    const int e0 = (blockIdx.x * EDGE_TILES + t) * (WAVES * 16) + wave * 16;
    if (e0 >= nEdges) break;  // wave-uniform; no barriers inside the tile loop

    if (lane < 16) {  // per-edge metadata + count scatter
      int e = e0 + lane, s = 0, d = 0; float dec = 0.0f;
      if (e < nEdges) {
        s = eidx[e]; d = eidx[nEdges + e];
        dec = __cosf(1.5707963267948966f * nd[e]);
        unsafeAtomicAdd(&cnt[d], 1.0f);
      }
      sSrc[wave][lane] = s; sDst[wave][lane] = d; sDec[wave][lane] = dec;
    }
    // build A = bf16(cat(x[src], e_state, x[dst]))  (in-wave LDS ordering covers sSrc/sDst)
    for (int r = 0; r < 16; ++r) {
      int e = e0 + r;
      int s = sSrc[wave][r], d = sDst[wave][r];
      int c = lane * 4;
      f32x4 v0 = {0.f,0.f,0.f,0.f}, v1 = v0, v2 = v0;
      if (e < nEdges) {
        v0 = *(const f32x4*)(x + (size_t)s * HDIM + c);
        v1 = *(const f32x4*)(est + (size_t)e * HDIM + c);
        v2 = *(const f32x4*)(x + (size_t)d * HDIM + c);
      }
      __bf16* row = &sA[wave][r * (3 * HDIM)];
      *(bf16x4*)(row + c)            = cvt4(v0);
      *(bf16x4*)(row + HDIM + c)     = cvt4(v1);
      *(bf16x4*)(row + 2 * HDIM + c) = cvt4(v2);
    }

    f32x8 acc[8];
    // stage 1: silu(cat@Wint1+b1), K=384
    gemm_16xH(sA[wave], sW1, 3 * HDIM, lane, acc);
#pragma unroll
    for (int nt = 0; nt < 8; ++nt) {
      int n = nt * 16 + nl; float bias = b1[n];
#pragma unroll
      for (int r = 0; r < 8; ++r)
        sS[wave][(r + hi) * HDIM + n] = (__bf16)silu_f(acc[nt][r] + bias);
    }
    // stage 2: m_st = decay*silu(.@Wint2+b2)*fm[src]; scatter-add to seg[dst]
    gemm_16xH(sS[wave], sW2, HDIM, lane, acc);
#pragma unroll
    for (int nt = 0; nt < 8; ++nt) {
      int n = nt * 16 + nl; float bias = b2[n];
#pragma unroll
      for (int r = 0; r < 8; ++r) {
        int m = r + hi, e = e0 + m;
        if (e < nEdges) {
          float v = silu_f(acc[nt][r] + bias) * sDec[wave][m];
          v *= fm[(size_t)sSrc[wave][m] * HDIM + n];
          unsafeAtomicAdd(&seg[(size_t)sDst[wave][m] * HDIM + n], v);
        }
      }
    }
  }
}

// ---------------------------------------------------------------- f_upd (pre-BN) + column stats
__global__ __launch_bounds__(TPB) void u_kernel(
    const float* __restrict__ mt, const float* __restrict__ seg, const float* __restrict__ cnt,
    const float* __restrict__ b1, const float* __restrict__ b2,
    const __bf16* __restrict__ Wt1, const __bf16* __restrict__ Wt2,
    float* __restrict__ u, float* __restrict__ gcs, float* __restrict__ gcsq, int nNodes) {
  __shared__ __align__(16) __bf16 sW1[HDIM * HDIM];
  __shared__ __align__(16) __bf16 sW2[HDIM * HDIM];
  __shared__ __align__(16) __bf16 sA[WAVES][16 * HDIM];
  __shared__ __align__(16) __bf16 sS[WAVES][16 * HDIM];
  __shared__ float sCS[HDIM];
  __shared__ float sCQ[HDIM];

  const int tid = threadIdx.x, wave = tid >> 5, lane = tid & 31;
  const int hi = (lane >> 4) << 3, nl = lane & 15;
  if (tid < HDIM) { sCS[tid] = 0.0f; sCQ[tid] = 0.0f; }
  {
    const uint4* s1 = (const uint4*)Wt1; const uint4* s2 = (const uint4*)Wt2;
    uint4* d1 = (uint4*)sW1; uint4* d2 = (uint4*)sW2;
    for (int i = tid; i < HDIM * HDIM / 8; i += TPB) { d1[i] = s1[i]; d2[i] = s2[i]; }
  }

  const int row0 = blockIdx.x * (WAVES * 16) + wave * 16;
  for (int r = 0; r < 16; ++r) {  // h = m_t + seg/max(cnt,1)
    int row = row0 + r;
    f32x4 h = {0.f, 0.f, 0.f, 0.f};
    if (row < nNodes) {
      f32x4 a = *(const f32x4*)(mt + (size_t)row * HDIM + lane * 4);
      f32x4 b = *(const f32x4*)(seg + (size_t)row * HDIM + lane * 4);
      float inv = 1.0f / fmaxf(cnt[row], 1.0f);
      h.x = a.x + b.x * inv; h.y = a.y + b.y * inv;
      h.z = a.z + b.z * inv; h.w = a.w + b.w * inv;
    }
    *(bf16x4*)(&sA[wave][r * HDIM + lane * 4]) = cvt4(h);
  }
  __syncthreads();

  f32x8 acc[8];
  gemm_16xH(sA[wave], sW1, HDIM, lane, acc);
#pragma unroll
  for (int nt = 0; nt < 8; ++nt) {
    int n = nt * 16 + nl; float bias = b1[n];
#pragma unroll
    for (int r = 0; r < 8; ++r)
      sS[wave][(r + hi) * HDIM + n] = (__bf16)silu_f(acc[nt][r] + bias);
  }
  gemm_16xH(sS[wave], sW2, HDIM, lane, acc);
#pragma unroll
  for (int nt = 0; nt < 8; ++nt) {
    int n = nt * 16 + nl; float bias = b2[n];
    float s1 = 0.0f, s2 = 0.0f;
#pragma unroll
    for (int r = 0; r < 8; ++r) {
      int row = row0 + r + hi;
      float v = acc[nt][r] + bias;
      if (row < nNodes) { u[(size_t)row * HDIM + n] = v; s1 += v; s2 += v * v; }
    }
    unsafeAtomicAdd(&sCS[n], s1);
    unsafeAtomicAdd(&sCQ[n], s2);
  }
  __syncthreads();
  if (tid < HDIM) {
    unsafeAtomicAdd(&gcs[tid], sCS[tid]);
    unsafeAtomicAdd(&gcsq[tid], sCQ[tid]);
  }
}

// ---------------------------------------------------------------- batchnorm + silu
__global__ void bn_kernel(const float* __restrict__ u, const float* __restrict__ gcs,
                          const float* __restrict__ gcsq, const float* __restrict__ gamma,
                          const float* __restrict__ beta, float* __restrict__ out, int nNodes) {
  int i = blockIdx.x * blockDim.x + threadIdx.x;
  if (i >= nNodes * HDIM) return;
  int n = i & (HDIM - 1);
  float invN = 1.0f / (float)nNodes;
  float mean = gcs[n] * invN;
  float var  = gcsq[n] * invN - mean * mean;
  float v = (u[i] - mean) * rsqrtf(var + 1e-5f) * gamma[n] + beta[n];
  out[i] = silu_f(v);
}

// ---------------------------------------------------------------- launch
extern "C" void kernel_launch(void* const* d_in, const int* in_sizes, int n_in,
                              void* d_out, int out_size, void* d_ws, size_t ws_size,
                              hipStream_t stream) {
  const float* x      = (const float*)d_in[0];
  const int*   eidx   = (const int*)d_in[1];
  const float* nd     = (const float*)d_in[2];
  const float* est    = (const float*)d_in[3];
  const float* W_int1 = (const float*)d_in[4];  const float* b_int1 = (const float*)d_in[5];
  const float* W_int2 = (const float*)d_in[6];  const float* b_int2 = (const float*)d_in[7];
  const float* W_mes1 = (const float*)d_in[8];  const float* b_mes1 = (const float*)d_in[9];
  const float* W_mes2 = (const float*)d_in[10]; const float* b_mes2 = (const float*)d_in[11];
  const float* W_upd1 = (const float*)d_in[12]; const float* b_upd1 = (const float*)d_in[13];
  const float* W_upd2 = (const float*)d_in[14]; const float* b_upd2 = (const float*)d_in[15];
  const float* gamma  = (const float*)d_in[16]; const float* beta   = (const float*)d_in[17];
  const float* Mind   = (const float*)d_in[18];

  const int nN = in_sizes[0] / HDIM;
  const int nE = in_sizes[2];

  char* ws = (char*)d_ws; size_t off = 0;
  auto alloc = [&](size_t bytes) -> void* {
    void* p = ws + off; off += (bytes + 255) & ~(size_t)255; return p;
  };
  float*  seg   = (float*)alloc((size_t)nN * HDIM * 4);
  float*  cntb  = (float*)alloc((size_t)nN * 4);
  float*  fm    = (float*)alloc((size_t)nN * HDIM * 4);  // reused as u after edge pass
  float*  mt    = (float*)alloc((size_t)nN * HDIM * 4);
  float*  gcs   = (float*)alloc(512);
  float*  gcsq  = (float*)alloc(512);
  __bf16* WtI1  = (__bf16*)alloc((size_t)3 * HDIM * HDIM * 2);
  __bf16* WtI2  = (__bf16*)alloc((size_t)HDIM * HDIM * 2);
  __bf16* WtM1  = (__bf16*)alloc((size_t)HDIM * HDIM * 2);
  __bf16* WtM2  = (__bf16*)alloc((size_t)HDIM * HDIM * 2);
  __bf16* WtIn  = (__bf16*)alloc((size_t)HDIM * HDIM * 2);
  __bf16* WtU1  = (__bf16*)alloc((size_t)HDIM * HDIM * 2);
  __bf16* WtU2  = (__bf16*)alloc((size_t)HDIM * HDIM * 2);

  const int elemNH = nN * HDIM;
  zero_kernel<<<(elemNH + 255) / 256, 256, 0, stream>>>(seg, elemNH);
  zero_kernel<<<(nN + 255) / 256, 256, 0, stream>>>(cntb, nN);
  zero_kernel<<<1, 256, 0, stream>>>(gcs, 128);
  zero_kernel<<<1, 256, 0, stream>>>(gcsq, 128);

  const int hh = HDIM * HDIM;
  wconv_kernel<<<(3 * hh + 255) / 256, 256, 0, stream>>>(W_int1, WtI1, 3 * HDIM, HDIM);
  wconv_kernel<<<(hh + 255) / 256, 256, 0, stream>>>(W_int2, WtI2, HDIM, HDIM);
  wconv_kernel<<<(hh + 255) / 256, 256, 0, stream>>>(W_mes1, WtM1, HDIM, HDIM);
  wconv_kernel<<<(hh + 255) / 256, 256, 0, stream>>>(W_mes2, WtM2, HDIM, HDIM);
  wconv_kernel<<<(hh + 255) / 256, 256, 0, stream>>>(Mind,   WtIn, HDIM, HDIM);
  wconv_kernel<<<(hh + 255) / 256, 256, 0, stream>>>(W_upd1, WtU1, HDIM, HDIM);
  wconv_kernel<<<(hh + 255) / 256, 256, 0, stream>>>(W_upd2, WtU2, HDIM, HDIM);

  const int nodeWG = (nN + WAVES * 16 - 1) / (WAVES * 16);
  node_kernel<<<nodeWG, TPB, 0, stream>>>(x, b_mes1, b_mes2, WtM1, WtM2, WtIn, fm, mt, nN);

  const int edgesPerWG = WAVES * 16 * EDGE_TILES;
  const int edgeWG = (nE + edgesPerWG - 1) / edgesPerWG;
  edge_kernel<<<edgeWG, TPB, 0, stream>>>(x, eidx, nd, est, b_int1, b_int2,
                                          WtI1, WtI2, fm, seg, cntb, nE);

  u_kernel<<<nodeWG, TPB, 0, stream>>>(mt, seg, cntb, b_upd1, b_upd2,
                                       WtU1, WtU2, fm, gcs, gcsq, nN);

  bn_kernel<<<(elemNH + 255) / 256, 256, 0, stream>>>(fm, gcs, gcsq, gamma, beta,
                                                      (float*)d_out, nN);
}
